// SigmoidImplicitLayer_7885559955905
// MI455X (gfx1250) — compile-verified, matrix operand order checked by reading, not run on hardware
//
#include <hip/hip_runtime.h>
#include <hip/hip_bf16.h>
#include <math.h>

// Problem constants (match reference)
#define BB   64
#define NN   512
#define DD   256
#define HH   512
#define NITER 20

typedef __attribute__((ext_vector_type(16))) __bf16 v16bf;
typedef __attribute__((ext_vector_type(8)))  __bf16 v8bf;
typedef __attribute__((ext_vector_type(8)))  float  v8f;

// fp32 -> bf16 round-to-nearest-even (bit pattern)
__device__ __forceinline__ unsigned short f2bf_bits(float f) {
    unsigned u = __builtin_bit_cast(unsigned, f);
    return (unsigned short)((u + 0x7FFFu + ((u >> 16) & 1u)) >> 16);
}
__device__ __forceinline__ __bf16 f2bf(float f) {
    unsigned short s = f2bf_bits(f);
    return __builtin_bit_cast(__bf16, s);
}

// Load one 16-bit WMMA fragment straight from global memory.
// ISA layout (16-bit A 16x32): lane group g holds K = [8g, 8g+8) in VGPRs 0-3
// and K = [16+8g, 16+8g+8) in VGPRs 4-7 -> two contiguous 16-byte row chunks.
// p points at (row, k_tile + 8*g); second chunk is +16 elements.
__device__ __forceinline__ v16bf ldfrag(const __bf16* p) {
    v8bf lo = *(const v8bf*)(p);
    v8bf hi = *(const v8bf*)(p + 16);
    return __builtin_shufflevector(lo, hi, 0, 1, 2, 3, 4, 5, 6, 7,
                                   8, 9, 10, 11, 12, 13, 14, 15);
}

__device__ __forceinline__ v8f wmma_bf16(v16bf a, v16bf b, v8f c) {
    return __builtin_amdgcn_wmma_f32_16x16x32_bf16(
        false, a, false, b, (short)0, c, false, false);
}

// ---------------------------------------------------------------------------
// One-time: V (fp32) -> Vbf (bf16), packed 4-at-a-time.
// ---------------------------------------------------------------------------
__global__ void __launch_bounds__(256)
convert_v_kernel(const float* __restrict__ V, __bf16* __restrict__ Vbf) {
    size_t i = ((size_t)blockIdx.x * 256 + threadIdx.x) * 4;
    float4 f = *(const float4*)(V + i);
    unsigned p0 = (unsigned)f2bf_bits(f.x) | ((unsigned)f2bf_bits(f.y) << 16);
    unsigned p1 = (unsigned)f2bf_bits(f.z) | ((unsigned)f2bf_bits(f.w) << 16);
    uint2 u; u.x = p0; u.y = p1;
    *(uint2*)(Vbf + i) = u;
}

// ---------------------------------------------------------------------------
// One-time: W1t[h][d] = bf16(W1[d][h]) for d < D (top half, transposed so the
// B-matrix fragments are contiguous-in-K too). Tiny (H*D = 128K elems).
// ---------------------------------------------------------------------------
__global__ void __launch_bounds__(256)
convert_w1t_kernel(const float* __restrict__ W1, __bf16* __restrict__ W1t) {
    int idx = blockIdx.x * 256 + threadIdx.x;   // over H*D
    int h = idx / DD;
    int d = idx - h * DD;
    W1t[idx] = f2bf(W1[(size_t)d * HH + h]);
}

// ---------------------------------------------------------------------------
// One-time WMMA GEMM: VW1[b,n,h] = sum_d Vbf[b,n,d] * W1t[h,d]
// 256 threads = 8 waves (2x4); each wave computes a 32x32 C tile with 2x2
// register blocking: 4 fragments loaded (2 b128 each) -> 4 WMMAs per K-step.
// No LDS, no barriers: fragments come straight from global (L1/L2 reuse).
// ---------------------------------------------------------------------------
__global__ void __launch_bounds__(256)
gemm_vw1_kernel(const __bf16* __restrict__ Vbf, const __bf16* __restrict__ W1t,
                float* __restrict__ vw1) {
    const int b   = blockIdx.z;
    const int tid = threadIdx.x;
    const int l   = tid & 31;
    const int w   = tid >> 5;     // wave 0..7
    const int wr  = w >> 2;       // 0..1
    const int wc  = w & 3;        // 0..3
    const int g   = l >> 4;       // lane group
    const int m   = l & 15;

    const int n0 = blockIdx.x * 64 + 32 * wr;    // this wave's 32 rows
    const int h0 = blockIdx.y * 128 + 32 * wc;   // this wave's 32 cols

    const __bf16* a0p = Vbf + ((size_t)b * NN + n0 + m) * DD + 8 * g;
    const __bf16* a1p = a0p + (size_t)16 * DD;
    const __bf16* b0p = W1t + (size_t)(h0 + m) * DD + 8 * g;
    const __bf16* b1p = b0p + (size_t)16 * DD;

    v8f acc00 = {}, acc01 = {}, acc10 = {}, acc11 = {};

    #pragma unroll
    for (int kt = 0; kt < DD; kt += 32) {
        v16bf a0 = ldfrag(a0p + kt);
        v16bf a1 = ldfrag(a1p + kt);
        v16bf b0 = ldfrag(b0p + kt);
        v16bf b1 = ldfrag(b1p + kt);
        acc00 = wmma_bf16(a0, b0, acc00);
        acc01 = wmma_bf16(a0, b1, acc01);
        acc10 = wmma_bf16(a1, b0, acc10);
        acc11 = wmma_bf16(a1, b1, acc11);
    }

    // C/D layout: VGPR r -> M = r + 8*g, N = lane%16
    float* cbase = vw1 + ((size_t)b * NN) * HH;
    #pragma unroll
    for (int i = 0; i < 2; ++i) {
        #pragma unroll
        for (int j = 0; j < 2; ++j) {
            const v8f acc = (i == 0) ? (j == 0 ? acc00 : acc01)
                                     : (j == 0 ? acc10 : acc11);
            float* p = cbase + (size_t)(n0 + 16 * i + 8 * g) * HH
                             + (h0 + 16 * j + m);
            #pragma unroll
            for (int r = 0; r < 8; ++r) p[(size_t)r * HH] = acc[r];
        }
    }
}

// ---------------------------------------------------------------------------
// Per iteration: pooled[b,:] = q[b,:]^T V[b], then
//                pw[b,h] = b1[h] + sum_d pooled[b,d]*W1[D+d,h]
// One block per batch, 512 threads.
// ---------------------------------------------------------------------------
__global__ void __launch_bounds__(512)
pool_pw_kernel(const float* __restrict__ q, const float* __restrict__ V,
               const float* __restrict__ W1, const float* __restrict__ b1,
               float* __restrict__ pw) {
    __shared__ float sq[NN];
    __shared__ float spart[512];
    __shared__ float spooled[DD];

    const int b   = blockIdx.x;
    const int tid = threadIdx.x;

    sq[tid] = q[(size_t)b * NN + tid];
    __syncthreads();

    {   // pooled: 2 threads per d (split n-range), coalesced over d
        const int d    = tid & (DD - 1);
        const int part = tid >> 8;
        const int nlo  = part * (NN / 2);
        float acc = 0.f;
        const float* vp = V + (size_t)b * NN * DD + (size_t)nlo * DD + d;
        #pragma unroll 4
        for (int n = 0; n < NN / 2; ++n) acc += sq[nlo + n] * vp[(size_t)n * DD];
        spart[tid] = acc;
    }
    __syncthreads();
    if (tid < DD) spooled[tid] = spart[tid] + spart[tid + DD];
    __syncthreads();

    {   // pw[b,h], coalesced over h
        const int h = tid;
        float acc = b1[h];
        #pragma unroll 4
        for (int d = 0; d < DD; ++d) acc += spooled[d] * W1[(size_t)(DD + d) * HH + h];
        pw[(size_t)b * HH + h] = acc;
    }
}

// ---------------------------------------------------------------------------
// Per iteration: q_out[b,n] = sigmoid(b2 + sum_h relu(VW1[b,n,h]+pw[b,h])*W2[h])
// One wave32 per (b,n) row, float4 streaming (L2-resident VW1), shuffle reduce.
// ---------------------------------------------------------------------------
__global__ void __launch_bounds__(256)
score_kernel(const float* __restrict__ vw1, const float* __restrict__ pw,
             const float* __restrict__ W2, const float* __restrict__ b2,
             float* __restrict__ q_out) {
    const int lane = threadIdx.x & 31;
    const int wv   = threadIdx.x >> 5;
    const int row  = blockIdx.x * 8 + wv;       // 0 .. B*N-1
    const int b    = row >> 9;                  // row / NN

    const float4* v4 = (const float4*)(vw1 + (size_t)row * HH);
    const float4* p4 = (const float4*)(pw + (size_t)b * HH);
    const float4* w4 = (const float4*)W2;

    float acc = 0.f;
    #pragma unroll
    for (int s = 0; s < HH / 128; ++s) {        // 4 steps of 32 lanes * 4 floats
        int i = lane + 32 * s;
        float4 v = v4[i], p = p4[i], wt = w4[i];
        float x0 = v.x + p.x, x1 = v.y + p.y, x2 = v.z + p.z, x3 = v.w + p.w;
        acc += (x0 > 0.f ? x0 : 0.f) * wt.x;
        acc += (x1 > 0.f ? x1 : 0.f) * wt.y;
        acc += (x2 > 0.f ? x2 : 0.f) * wt.z;
        acc += (x3 > 0.f ? x3 : 0.f) * wt.w;
    }
    #pragma unroll
    for (int off = 16; off > 0; off >>= 1) acc += __shfl_down(acc, off, 32);
    if (lane == 0) {
        float s = acc + b2[0];
        q_out[row] = 1.f / (1.f + expf(-s));
    }
}

// ---------------------------------------------------------------------------
extern "C" void kernel_launch(void* const* d_in, const int* in_sizes, int n_in,
                              void* d_out, int out_size, void* d_ws, size_t ws_size,
                              hipStream_t stream) {
    const float* q_init = (const float*)d_in[0];
    const float* V      = (const float*)d_in[1];
    const float* W1     = (const float*)d_in[2];
    const float* b1     = (const float*)d_in[3];
    const float* W2     = (const float*)d_in[4];
    const float* b2     = (const float*)d_in[5];
    float* out = (float*)d_out;

    // Workspace carve-up
    float*  vw1 = (float*)d_ws;                         // B*N*H fp32 (67 MB)
    float*  pw  = vw1 + (size_t)BB * NN * HH;           // B*H
    float*  qA  = pw + (size_t)BB * HH;                 // B*N
    float*  qB  = qA + (size_t)BB * NN;                 // B*N
    __bf16* Vbf = (__bf16*)(qB + (size_t)BB * NN);      // B*N*D bf16 (16.7 MB)
    __bf16* W1t = Vbf + (size_t)BB * NN * DD;           // H*D bf16 (0.25 MB)

    // One-time precompute of the iteration-invariant V @ W1[:D]
    convert_v_kernel<<<(BB * NN * DD) / (256 * 4), 256, 0, stream>>>(V, Vbf);
    convert_w1t_kernel<<<(HH * DD) / 256, 256, 0, stream>>>(W1, W1t);
    {
        dim3 grid(NN / 64, HH / 128, BB);
        gemm_vw1_kernel<<<grid, 256, 0, stream>>>(Vbf, W1t, vw1);
    }

    // 20 fixed-point iterations
    for (int it = 0; it < NITER; ++it) {
        const float* q_in = (it == 0) ? q_init : ((it & 1) ? qA : qB);
        float* q_nx = (it == NITER - 1) ? out : ((it & 1) ? qB : qA);

        pool_pw_kernel<<<BB, 512, 0, stream>>>(q_in, V, W1, b1, pw);
        score_kernel<<<(BB * NN) / 8, 256, 0, stream>>>(vw1, pw, W2, b2, q_nx);
    }
}
